// MoE_Bottleneck_35476429865769
// MI455X (gfx1250) — matrix operand order repaired and checked by reading, not compiled, not run on hardware
//
#include <hip/hip_runtime.h>

typedef __attribute__((ext_vector_type(16))) _Float16 v16h;
typedef __attribute__((ext_vector_type(8)))  _Float16 v8h;
typedef __attribute__((ext_vector_type(8)))  float    v8f;
typedef __attribute__((ext_vector_type(4)))  float    v4f;

#define TOKENS 25600   // 160*160
#define C1     512
#define CMID   256
#define C2     512

__device__ __forceinline__ float silu_f(float v) {
    return v / (1.0f + __expf(-v));
}

// ---------------------------------------------------------------------------
// CDNA5 async global->LDS copy (ASYNCcnt-tracked, no VGPR data path).
// LDS address = low 32 bits of the generic pointer (ISA 10.2: LDS aperture
// keeps addr[31:0]). 16B alignment guaranteed by our staging layout.
// ---------------------------------------------------------------------------
__device__ __forceinline__ void async_ld128(void* lds, const void* g) {
    unsigned ldsOff = (unsigned)(unsigned long long)lds;
    asm volatile("global_load_async_to_lds_b128 %0, %1, off"
                 :: "v"(ldsOff), "v"(g) : "memory");
}
__device__ __forceinline__ void wait_async0() {
    asm volatile("s_wait_asynccnt 0x0" ::: "memory");
}

// ---------------------------------------------------------------------------
// Pack weights into per-lane WMMA B-fragment layout (f16).
// Lane half h: elems 0-7 hold K=kt*32+8h+0..7, elems 8-15 hold K=kt*32+16+8h+0..7;
// column = nt*16+lane%16. Frag order f = kt*NT + nt.
//   region 0: GEMM1 (w1^T as [512 x 256]):  KT=16, NT=16 -> 256 frags
//   region 1: experts e=0..3 (wc_e^T [256 x 256]): KT=8, NT=16 -> 4*128 frags
//   region 2: GEMM3 (w3^T [256 x 512]): KT=8, NT=32 -> 256 frags
// ---------------------------------------------------------------------------
__global__ void pack_weights_k(const float* __restrict__ w1,
                               const float* __restrict__ wexp,
                               const float* __restrict__ w3,
                               _Float16* __restrict__ pb1,
                               _Float16* __restrict__ pbe,
                               _Float16* __restrict__ pb3) {
    int gid  = blockIdx.x * blockDim.x + threadIdx.x;   // 0..32767
    int lane = gid & 31;
    int f    = gid >> 5;                                // 0..1023
    int laneN = lane & 15;
    int kHalf = (lane >> 4) << 4;                       // 0 or 16
    int kt, nt;
    const float* W = w1; long ldW = 512; bool isExp = false; long eOff = 0;
    _Float16* out;
    if (f < 256) {
        kt = f >> 4; nt = f & 15; W = w1; ldW = 512;
        out = pb1 + ((long)f * 32 + lane) * 16;
    } else if (f < 768) {
        int fe = f - 256; int e = fe >> 7; int fl = fe & 127;
        kt = fl >> 4; nt = fl & 15; isExp = true; eOff = (long)e * 256;
        out = pbe + ((long)fe * 32 + lane) * 16;
    } else {
        int fg = f - 768; kt = fg >> 5; nt = fg & 31; W = w3; ldW = 256;
        out = pb3 + ((long)fg * 32 + lane) * 16;
    }
    int col = nt * 16 + laneN;
    v8h lo, hi;
#pragma unroll
    for (int m = 0; m < 16; ++m) {
        int K = kt * 32 + ((lane >> 4) << 3) + (m & 7) + ((m >> 3) << 4);
        float v;
        if (isExp) {
            v = wexp[(((eOff + col) * 256L + K) * 9L) + 4];  // center tap [1][1]
        } else {
            v = W[(long)col * ldW + K];
        }
        if (m < 8) lo[m] = (_Float16)v; else hi[m - 8] = (_Float16)v;
    }
    *(v8h*)out       = lo;
    *(v8h*)(out + 8) = hi;
}

// ---------------------------------------------------------------------------
// x [512][25600] f32  ->  xT [25600][512] f16  (tiled transpose via LDS)
// ---------------------------------------------------------------------------
__global__ __launch_bounds__(256) void transpose_cast_k(const float* __restrict__ x,
                                                        _Float16* __restrict__ xT) {
    __shared__ float tile[64][65];
    int tBase = blockIdx.x * 64;
    int cBase = blockIdx.y * 64;
    int tid = threadIdx.x;
    {
        int r = tid >> 2;            // channel row 0..63
        int q = tid & 3;             // 16-float segment
        const float* src = x + (long)(cBase + r) * TOKENS + tBase + q * 16;
#pragma unroll
        for (int k = 0; k < 4; ++k) {
            v4f v = *(const v4f*)(src + k * 4);
            *(v4f*)&tile[r][q * 16 + k * 4] = v;
        }
    }
    __syncthreads();
    {
        int t   = tid >> 2;          // token row 0..63
        int seg = tid & 3;           // 16-half segment
        v8h a, b;
#pragma unroll
        for (int k = 0; k < 8; ++k)  a[k] = (_Float16)tile[seg * 16 + k][t];
#pragma unroll
        for (int k = 0; k < 8; ++k)  b[k] = (_Float16)tile[seg * 16 + 8 + k][t];
        _Float16* dst = xT + (long)(tBase + t) * C1 + cBase + seg * 16;
        *(v8h*)dst       = a;
        *(v8h*)(dst + 8) = b;
    }
}

// ---------------------------------------------------------------------------
// GEMM1 + BN1 + SiLU + router (softmax/top-2/renorm -> gate[T][4])
// block: 128 tokens x 256 channels, K=512. 8 waves: 2(m) x 4(n), wave 64x64.
// Double-buffered async A staging: buffers at smem[0] / smem[5120].
// ---------------------------------------------------------------------------
__global__ __launch_bounds__(256) void gemm1_router_k(
        const _Float16* __restrict__ xT,        // [T][512]
        const _Float16* __restrict__ pb1,
        const float* __restrict__ bn1_scale, const float* __restrict__ bn1_shift,
        const float* __restrict__ w_router,     // [4][256]
        const float* __restrict__ b_router,     // [4]
        _Float16* __restrict__ x1,              // [T][256]
        float* __restrict__ gate)               // [T][4]
{
    __shared__ alignas(16) _Float16 smem[32768]; // 64KB union: 2x A-stage [128][40] / x1 tile [128][256] swizzled
    _Float16* X1s = smem;

    int tid  = threadIdx.x;
    int lane = tid & 31;
    int wv   = tid >> 5;
    int mw   = wv >> 2;          // 0..1
    int nw   = wv & 3;           // 0..3
    int laneN  = lane & 15;
    int laneHi = lane >> 4;      // 0/1
    long tBase = (long)blockIdx.x * 128;

    int rr = tid >> 1;           // staging row 0..127
    int hs = tid & 1;            // staging 16-half segment
    const _Float16* aBase = xT + (tBase + rr) * C1 + hs * 16;
    int ldsOffH = rr * 40 + hs * 16;

    // prefetch K-step 0 into buffer 0
    async_ld128(&smem[ldsOffH],     aBase);
    async_ld128(&smem[ldsOffH + 8], aBase + 8);

    v8f acc[4][4] = {};

    for (int ks = 0; ks < 16; ++ks) {
        wait_async0();
        __syncthreads();
        if (ks + 1 < 16) {       // prefetch next K-step into the other buffer
            _Float16* d = smem + ((ks + 1) & 1) * 5120;
            const _Float16* gs = aBase + (ks + 1) * 32;
            async_ld128(&d[ldsOffH],     gs);
            async_ld128(&d[ldsOffH + 8], gs + 8);
        }
        _Float16* As = smem + (ks & 1) * 5120;
        v16h afrag[4];
#pragma unroll
        for (int mt = 0; mt < 4; ++mt) {
            int row = mw * 64 + mt * 16 + laneN;
            const _Float16* ap = &As[row * 40 + laneHi * 8];
            v8h lo = *(const v8h*)ap;
            v8h hi = *(const v8h*)(ap + 16);
            afrag[mt] = __builtin_shufflevector(lo, hi, 0,1,2,3,4,5,6,7,8,9,10,11,12,13,14,15);
        }
#pragma unroll
        for (int nt = 0; nt < 4; ++nt) {
            int ntG = nw * 4 + nt;
            v16h b = *(const v16h*)(pb1 + (((long)ks * 16 + ntG) * 32 + lane) * 16);
#pragma unroll
            for (int mt = 0; mt < 4; ++mt)
                acc[mt][nt] = __builtin_amdgcn_wmma_f32_16x16x32_f16(
                    false, afrag[mt], false, b, (short)0, acc[mt][nt], false, false);
        }
    }
    __syncthreads();

    // epilogue: BN + SiLU -> swizzled f16 tile in LDS (row=token, 32 chunks of 8 halves)
#pragma unroll
    for (int nt = 0; nt < 4; ++nt) {
        int c = nw * 64 + nt * 16 + laneN;
        float sc = bn1_scale[c], sh = bn1_shift[c];
#pragma unroll
        for (int mt = 0; mt < 4; ++mt) {
#pragma unroll
            for (int i = 0; i < 8; ++i) {
                float v = acc[mt][nt][i] * sc + sh;
                float s = silu_f(v);
                int r = mw * 64 + mt * 16 + laneHi * 8 + i;
                int pch = (c >> 3) ^ (r & 31);
                X1s[r * 256 + pch * 8 + (c & 7)] = (_Float16)s;
            }
        }
    }
    __syncthreads();

    // coalesced global store of x1 tile
    {
        _Float16* dst = x1 + (tBase + rr) * 256 + hs * 128;
#pragma unroll
        for (int cc = 0; cc < 16; ++cc) {
            int chunk = hs * 16 + cc;
            int pch = chunk ^ (rr & 31);
            v8h v = *(v8h*)&X1s[rr * 256 + pch * 8];
            *(v8h*)(dst + cc * 8) = v;
        }
    }

    // router: 128 threads, one token each (conflict-free via swizzle)
    if (tid < 128) {
        int r = tid;
        float l0 = b_router[0], l1 = b_router[1], l2 = b_router[2], l3 = b_router[3];
        for (int chunk = 0; chunk < 32; ++chunk) {
            int pch = chunk ^ (r & 31);
            v8h v = *(v8h*)&X1s[r * 256 + pch * 8];
#pragma unroll
            for (int q = 0; q < 8; ++q) {
                float xv = (float)v[q];
                int c = chunk * 8 + q;
                l0 += xv * w_router[c];
                l1 += xv * w_router[256 + c];
                l2 += xv * w_router[512 + c];
                l3 += xv * w_router[768 + c];
            }
        }
        float l[4] = { l0, l1, l2, l3 };
        int i1 = 0; float m1 = l[0];
#pragma unroll
        for (int e = 1; e < 4; ++e) if (l[e] > m1) { m1 = l[e]; i1 = e; }
        int i2 = -1; float m2 = -3.4e38f;
#pragma unroll
        for (int e = 0; e < 4; ++e) if (e != i1 && l[e] > m2) { m2 = l[e]; i2 = e; }
        float p2  = __expf(m2 - m1);
        float inv = 1.0f / (1.0f + p2);     // gate(top1)=1/(1+p2), gate(top2)=p2/(1+p2)
        v4f gv;
#pragma unroll
        for (int e = 0; e < 4; ++e)
            gv[e] = (e == i1) ? inv : ((e == i2) ? p2 * inv : 0.0f);
        *(v4f*)&gate[(tBase + r) * 4] = gv;
    }
}

// ---------------------------------------------------------------------------
// Experts: loop e=0..3, y=x1@wc_e^T, BN_e+SiLU, *gate[t][e], sum -> moe [T][256]
// block: 128 tokens x 128 channels, grid.y=2. waves 2(m) x 4(n), wave 64x32.
// A tile depends only on ks -> flattened (e,ks) double-buffered async prefetch.
// ---------------------------------------------------------------------------
__global__ __launch_bounds__(256) void gemm_experts_k(
        const _Float16* __restrict__ x1,        // [T][256]
        const _Float16* __restrict__ pbe,
        const float* __restrict__ bn_e_scale, const float* __restrict__ bn_e_shift, // [4][256]
        const float* __restrict__ gate,         // [T][4]
        _Float16* __restrict__ moe)             // [T][256]
{
    __shared__ alignas(16) _Float16 smem[16384]; // 32KB union: 2x A-stage [128][40] / out tile [128][128] swizzled
    __shared__ float gtile[512];                 // [128][4]

    int tid  = threadIdx.x;
    int lane = tid & 31;
    int wv   = tid >> 5;
    int mw   = wv >> 2;
    int nw   = wv & 3;
    int laneN  = lane & 15;
    int laneHi = lane >> 4;
    long tBase = (long)blockIdx.x * 128;
    int  cBase = blockIdx.y * 128;

    if (tid < 128)
        *(v4f*)&gtile[tid * 4] = *(const v4f*)&gate[(tBase + tid) * 4];

    int rr = tid >> 1;
    int hs = tid & 1;
    const _Float16* aBase = x1 + (tBase + rr) * 256 + hs * 16;
    int ldsOffH = rr * 40 + hs * 16;

    // prefetch step 0 into buffer 0
    async_ld128(&smem[ldsOffH],     aBase);
    async_ld128(&smem[ldsOffH + 8], aBase + 8);

    v8f oacc[4][2] = {};

    for (int e = 0; e < 4; ++e) {
        v8f yacc[4][2] = {};
        for (int ks = 0; ks < 8; ++ks) {
            int step = e * 8 + ks;
            wait_async0();
            __syncthreads();
            if (step + 1 < 32) {
                int nks = (step + 1) & 7;
                _Float16* d = smem + ((step + 1) & 1) * 5120;
                const _Float16* gs = aBase + nks * 32;
                async_ld128(&d[ldsOffH],     gs);
                async_ld128(&d[ldsOffH + 8], gs + 8);
            }
            _Float16* As = smem + (step & 1) * 5120;
            v16h afrag[4];
#pragma unroll
            for (int mt = 0; mt < 4; ++mt) {
                int row = mw * 64 + mt * 16 + laneN;
                const _Float16* ap = &As[row * 40 + laneHi * 8];
                v8h lo = *(const v8h*)ap;
                v8h hi = *(const v8h*)(ap + 16);
                afrag[mt] = __builtin_shufflevector(lo, hi, 0,1,2,3,4,5,6,7,8,9,10,11,12,13,14,15);
            }
#pragma unroll
            for (int nt = 0; nt < 2; ++nt) {
                int ntG = blockIdx.y * 8 + nw * 2 + nt;   // 0..15
                v16h b = *(const v16h*)(pbe + (((long)e * 128 + ks * 16 + ntG) * 32 + lane) * 16);
#pragma unroll
                for (int mt = 0; mt < 4; ++mt)
                    yacc[mt][nt] = __builtin_amdgcn_wmma_f32_16x16x32_f16(
                        false, afrag[mt], false, b, (short)0, yacc[mt][nt], false, false);
            }
        }
        // per-expert epilogue (registers only)
#pragma unroll
        for (int nt = 0; nt < 2; ++nt) {
            int c = cBase + nw * 32 + nt * 16 + laneN;
            float sc = bn_e_scale[e * 256 + c], sh = bn_e_shift[e * 256 + c];
#pragma unroll
            for (int mt = 0; mt < 4; ++mt) {
#pragma unroll
                for (int i = 0; i < 8; ++i) {
                    int r = mw * 64 + mt * 16 + laneHi * 8 + i;
                    float v = yacc[mt][nt][i] * sc + sh;
                    oacc[mt][nt][i] += silu_f(v) * gtile[r * 4 + e];
                }
            }
        }
    }
    __syncthreads();
    // repack: swizzled f16 out tile [128][128] (16 chunks/row)
#pragma unroll
    for (int nt = 0; nt < 2; ++nt) {
        int cl = nw * 32 + nt * 16 + laneN;
#pragma unroll
        for (int mt = 0; mt < 4; ++mt) {
#pragma unroll
            for (int i = 0; i < 8; ++i) {
                int r = mw * 64 + mt * 16 + laneHi * 8 + i;
                int pch = (cl >> 3) ^ (r & 15);
                smem[r * 128 + pch * 8 + (cl & 7)] = (_Float16)oacc[mt][nt][i];
            }
        }
    }
    __syncthreads();
    {
        _Float16* dst = moe + (tBase + rr) * 256 + cBase + hs * 64;
#pragma unroll
        for (int cc = 0; cc < 8; ++cc) {
            int chunk = hs * 8 + cc;
            int pch = chunk ^ (rr & 15);
            v8h v = *(v8h*)&smem[rr * 128 + pch * 8];
            *(v8h*)(dst + cc * 8) = v;
        }
    }
}

// ---------------------------------------------------------------------------
// GEMM3 + BN3 + SiLU + residual, output fp32 channel-major [512][25600].
// block: 128 tokens x 128 out-channels, grid.y=4. LDS f32 transpose tile for
// coalesced channel-major stores + residual reads.
// ---------------------------------------------------------------------------
__global__ __launch_bounds__(256) void gemm3_out_k(
        const _Float16* __restrict__ moe,       // [T][256]
        const _Float16* __restrict__ pb3,
        const float* __restrict__ bn3_scale, const float* __restrict__ bn3_shift,
        const float* __restrict__ x,            // [512][T]
        float* __restrict__ out)                // [512][T]
{
    __shared__ alignas(16) float fsm[16384];    // 64KB union: 2x A-stage (front 20KB) / f32 tile [128][128] swizzled
    _Float16* hsm = (_Float16*)fsm;

    int tid  = threadIdx.x;
    int lane = tid & 31;
    int wv   = tid >> 5;
    int mw   = wv >> 2;
    int nw   = wv & 3;
    int laneN  = lane & 15;
    int laneHi = lane >> 4;
    long tBase = (long)blockIdx.x * 128;
    int  oBase = blockIdx.y * 128;

    int rr = tid >> 1;
    int hs = tid & 1;
    const _Float16* aBase = moe + (tBase + rr) * 256 + hs * 16;
    int ldsOffH = rr * 40 + hs * 16;

    async_ld128(&hsm[ldsOffH],     aBase);
    async_ld128(&hsm[ldsOffH + 8], aBase + 8);

    v8f acc[4][2] = {};

    for (int ks = 0; ks < 8; ++ks) {
        wait_async0();
        __syncthreads();
        if (ks + 1 < 8) {
            _Float16* d = hsm + ((ks + 1) & 1) * 5120;
            const _Float16* gs = aBase + (ks + 1) * 32;
            async_ld128(&d[ldsOffH],     gs);
            async_ld128(&d[ldsOffH + 8], gs + 8);
        }
        _Float16* As = hsm + (ks & 1) * 5120;
        v16h afrag[4];
#pragma unroll
        for (int mt = 0; mt < 4; ++mt) {
            int row = mw * 64 + mt * 16 + laneN;
            const _Float16* ap = &As[row * 40 + laneHi * 8];
            v8h lo = *(const v8h*)ap;
            v8h hi = *(const v8h*)(ap + 16);
            afrag[mt] = __builtin_shufflevector(lo, hi, 0,1,2,3,4,5,6,7,8,9,10,11,12,13,14,15);
        }
#pragma unroll
        for (int nt = 0; nt < 2; ++nt) {
            int ntG = blockIdx.y * 8 + nw * 2 + nt;   // 0..31
            v16h b = *(const v16h*)(pb3 + (((long)ks * 32 + ntG) * 32 + lane) * 16);
#pragma unroll
            for (int mt = 0; mt < 4; ++mt)
                acc[mt][nt] = __builtin_amdgcn_wmma_f32_16x16x32_f16(
                    false, afrag[mt], false, b, (short)0, acc[mt][nt], false, false);
        }
    }
    __syncthreads();
    // BN + SiLU -> swizzled f32 tile (row=token, 32 chunks of 4 floats)
#pragma unroll
    for (int nt = 0; nt < 2; ++nt) {
        int cl = nw * 32 + nt * 16 + laneN;          // local out-channel 0..127
        int oc = oBase + cl;
        float sc = bn3_scale[oc], sh = bn3_shift[oc];
#pragma unroll
        for (int mt = 0; mt < 4; ++mt) {
#pragma unroll
            for (int i = 0; i < 8; ++i) {
                int r = mw * 64 + mt * 16 + laneHi * 8 + i;
                float v = acc[mt][nt][i] * sc + sh;
                int pch = (cl >> 2) ^ (r & 31);
                fsm[r * 128 + pch * 4 + (cl & 3)] = silu_f(v);
            }
        }
    }
    __syncthreads();
    // transposed, coalesced channel-major store + residual
    {
        int o    = tid >> 1;                 // local out-channel 0..127
        int tOff = (tid & 1) * 64;
        const float* xr   = x   + (long)(oBase + o) * TOKENS + tBase + tOff;
        float*       orow = out + (long)(oBase + o) * TOKENS + tBase + tOff;
#pragma unroll
        for (int t4 = 0; t4 < 16; ++t4) {
            v4f res = *(const v4f*)(xr + t4 * 4);
            v4f ov;
#pragma unroll
            for (int k = 0; k < 4; ++k) {
                int r = tOff + t4 * 4 + k;
                int pch = (o >> 2) ^ (r & 31);
                ov[k] = fsm[r * 128 + pch * 4 + (o & 3)] + res[k];
            }
            *(v4f*)(orow + t4 * 4) = ov;
        }
    }
}

// ---------------------------------------------------------------------------
// Workspace layout (bytes), total ~40.8 MB:
//   xT  @ 0          : 26,214,400  (aliased by moe after kernel1 is done)
//   x1  @ 26,214,400 : 13,107,200
//   gate@ 39,321,600 :    409,600
//   pb1 @ 39,731,200 :    262,144
//   pbe @ 39,993,344 :    524,288
//   pb3 @ 40,517,632 :    262,144
// ---------------------------------------------------------------------------
extern "C" void kernel_launch(void* const* d_in, const int* in_sizes, int n_in,
                              void* d_out, int out_size, void* d_ws, size_t ws_size,
                              hipStream_t stream) {
    (void)in_sizes; (void)n_in; (void)out_size; (void)ws_size;
    const float* x          = (const float*)d_in[0];
    const float* w1         = (const float*)d_in[1];
    const float* bn1_scale  = (const float*)d_in[2];
    const float* bn1_shift  = (const float*)d_in[3];
    const float* w_router   = (const float*)d_in[4];
    const float* b_router   = (const float*)d_in[5];
    const float* w_experts  = (const float*)d_in[6];
    const float* bn_e_scale = (const float*)d_in[7];
    const float* bn_e_shift = (const float*)d_in[8];
    const float* w3         = (const float*)d_in[9];
    const float* bn3_scale  = (const float*)d_in[10];
    const float* bn3_shift  = (const float*)d_in[11];

    char* ws = (char*)d_ws;
    _Float16* xT   = (_Float16*)(ws + 0);
    _Float16* moe  = (_Float16*)(ws + 0);            // alias: xT dead after gemm1
    _Float16* x1   = (_Float16*)(ws + 26214400);
    float*    gate = (float*)   (ws + 39321600);
    _Float16* pb1  = (_Float16*)(ws + 39731200);
    _Float16* pbe  = (_Float16*)(ws + 39993344);
    _Float16* pb3  = (_Float16*)(ws + 40517632);

    pack_weights_k<<<128, 256, 0, stream>>>(w1, w_experts, w3, pb1, pbe, pb3);
    transpose_cast_k<<<dim3(TOKENS / 64, C1 / 64), 256, 0, stream>>>(x, xT);
    gemm1_router_k<<<TOKENS / 128, 256, 0, stream>>>(xT, pb1, bn1_scale, bn1_shift,
                                                     w_router, b_router, x1, gate);
    gemm_experts_k<<<dim3(TOKENS / 128, 2), 256, 0, stream>>>(x1, pbe, bn_e_scale,
                                                              bn_e_shift, gate, moe);
    gemm3_out_k<<<dim3(TOKENS / 128, 4), 256, 0, stream>>>(moe, pb3, bn3_scale,
                                                           bn3_shift, x, (float*)d_out);
}
